// GenerativeNet_72962904424948
// MI455X (gfx1250) — matrix operand order repaired and checked by reading, not compiled, not run on hardware
//
#include <hip/hip_runtime.h>
#include <cstdint>

typedef __attribute__((ext_vector_type(16))) _Float16 v16h;
typedef __attribute__((ext_vector_type(8)))  float    v8f;

#define BN_EPS 1e-5f

__device__ __forceinline__ float elu1(float x) {
    return x > 0.f ? x : (__expf(x) - 1.f);
}

// ---------------------------------------------------------------------------
// Weight converter: src f32 [Co][Ci][T] (taps innermost)  ->  dst f16 [T][Co][CiP]
// K (ci) zero-padded to CiP (multiple of 32) so WMMA K-loops need no tail.
// ---------------------------------------------------------------------------
__global__ void k_cvt_w(const float* __restrict__ src, _Float16* __restrict__ dst,
                        int Co, int Ci, int CiP, int T) {
    long total = (long)T * Co * CiP;
    for (long i = blockIdx.x * (long)blockDim.x + threadIdx.x; i < total;
         i += (long)gridDim.x * blockDim.x) {
        int cip = (int)(i % CiP);
        long r = i / CiP;
        int co = (int)(r % Co);
        int t  = (int)(r / Co);
        float v = (cip < Ci) ? src[((long)co * Ci + cip) * T + t] : 0.f;
        dst[i] = (_Float16)v;
    }
}

__global__ void k_cvt_act(const float* __restrict__ src, _Float16* __restrict__ dst, long n) {
    long i = blockIdx.x * (long)blockDim.x + threadIdx.x;
    if (i < n) dst[i] = (_Float16)src[i];
}

// ---------------------------------------------------------------------------
// A fragment: weights row-major [M][ldk]; lane m = lane&15, g = lane>>4.
// 16-bit A 16x32 layout: VGPR j<4 -> K = 2j+8g, VGPR j>=4 -> K = 16+2(j-4)+8g.
// ---------------------------------------------------------------------------
__device__ __forceinline__ v16h load_a_frag(const _Float16* __restrict__ row, int g, int kb) {
    v16h a;
#pragma unroll
    for (int j = 0; j < 8; ++j) {
        int ko = (j < 4) ? (2 * j + 8 * g) : (16 + 2 * (j - 4) + 8 * g);
        a[2 * j]     = row[kb + ko];
        a[2 * j + 1] = row[kb + ko + 1];
    }
    return a;
}

// B fragment from LDS stored transposed [n][k]: lane n = lane&15, g = lane>>4.
// 16-bit B 32x16: element j<8 -> K = 8g+j, element j>=8 -> K = 16+8g+(j-8) (matches A).
__device__ __forceinline__ v16h load_b_frag(const _Float16* __restrict__ col, int g, int kb) {
    const _Float16* p = col + kb + 8 * g;
    v16h b;
#pragma unroll
    for (int j = 0; j < 16; ++j) b[j] = p[(j < 8) ? j : (j + 8)];
    return b;
}

// ---------------------------------------------------------------------------
// Transposed conv k=2 s=2 as 8 parity GEMMs over INPUT voxels.
// act f16 [Ci][Nin], W f16 [8][Co][CiP].  Epilogue: BN + ELU, scatter to [Co][Nout].
// Block = 256 threads = 8 waves = mWaves x nWaves tiles. LDS: x tile transposed.
// ---------------------------------------------------------------------------
__global__ void __launch_bounds__(256)
k_deconv(const _Float16* __restrict__ act, const _Float16* __restrict__ W,
         const float* __restrict__ bnG, const float* __restrict__ bnB,
         const float* __restrict__ bnM, const float* __restrict__ bnV,
         _Float16* __restrict__ out,
         int Ci, int CiP, int Co, int Din, int Hin, int Win,
         int mWaves, int nWaves) {
    extern __shared__ _Float16 sm[];
    const int ldk = CiP + 8;
    const int tid = threadIdx.x;
    const int Nin = Din * Hin * Win;
    const int nSlots = nWaves * 16;
    const int blockVox = blockIdx.x * nSlots;

    // stage activation tile transposed: sm[slot][k]
    for (int i = tid; i < nSlots * CiP; i += blockDim.x) {
        int s = i % nSlots;
        int cip = i / nSlots;
        int vox = blockVox + s;
        _Float16 val = (_Float16)0.f;
        if (cip < Ci && vox < Nin) val = act[(long)cip * Nin + vox];
        sm[s * ldk + cip] = val;
    }
    __syncthreads();

    const int wave = tid >> 5, lane = tid & 31;
    const int mw = wave % mWaves, nw = wave / mWaves;
    const int mTile = blockIdx.y * mWaves + mw;
    if (mTile * 16 >= Co) return;           // no barriers after this point

    const int n = lane & 15, g = lane >> 4;
    const int vox = blockVox + nw * 16 + n;
    const _Float16* bRow = sm + (nw * 16 + n) * ldk;

    float scale[8], shift[8];
#pragma unroll
    for (int r = 0; r < 8; ++r) {
        int co = mTile * 16 + r + 8 * g;
        float sc = bnG[co] * rsqrtf(bnV[co] + BN_EPS);
        scale[r] = sc;
        shift[r] = bnB[co] - bnM[co] * sc;
    }
    bool valid = vox < Nin;
    int wi = 0, hi = 0, di = 0;
    if (valid) { wi = vox % Win; hi = (vox / Win) % Hin; di = vox / (Win * Hin); }
    const int Wout = Win * 2, Hout = Hin * 2;
    const long Nout = (long)Nin * 8;

    for (int p = 0; p < 8; ++p) {
        const _Float16* aRow = W + ((long)(p * Co + mTile * 16 + n)) * CiP;
        v8f acc = {};
        for (int kb = 0; kb < CiP; kb += 32) {
            v16h a = load_a_frag(aRow, g, kb);
            v16h b = load_b_frag(bRow, g, kb);
            acc = __builtin_amdgcn_wmma_f32_16x16x32_f16(false, a, false, b,
                                                         (short)0, acc, false, false);
        }
        if (valid) {
            int pd = p >> 2, ph = (p >> 1) & 1, pw = p & 1;
            long od = ((long)(2 * di + pd) * Hout + (2 * hi + ph)) * Wout + (2 * wi + pw);
#pragma unroll
            for (int r = 0; r < 8; ++r) {
                int co = mTile * 16 + r + 8 * g;
                float x = acc[r] * scale[r] + shift[r];
                out[(long)co * Nout + od] = (_Float16)elu1(x);
            }
        }
    }
}

// ---------------------------------------------------------------------------
// 3x3x3 SAME conv as 27 shifted 1x1 GEMMs over OUTPUT voxels (one accumulator).
// act f16 [Ci][N], W f16 [27][Co][CiP]. Epilogue: BN + ELU + parent support mask.
// ---------------------------------------------------------------------------
__global__ void __launch_bounds__(256)
k_conv3(const _Float16* __restrict__ act, const _Float16* __restrict__ W,
        const float* __restrict__ bnG, const float* __restrict__ bnB,
        const float* __restrict__ bnM, const float* __restrict__ bnV,
        const unsigned char* __restrict__ sup,   // keep mask at (D/2)^3, or null
        _Float16* __restrict__ out,
        int Ci, int CiP, int Co, int D, int H, int Wd,
        int mWaves, int nWaves) {
    extern __shared__ _Float16 sm[];
    __shared__ int s_src[64];
    const int ldk = CiP + 8;
    const int tid = threadIdx.x;
    const int N = D * H * Wd;
    const int nSlots = nWaves * 16;
    const int blockVox = blockIdx.x * nSlots;
    const int wave = tid >> 5, lane = tid & 31;
    const int mw = wave % mWaves, nw = wave / mWaves;
    const int mTile = blockIdx.y * mWaves + mw;
    const int n = lane & 15, g = lane >> 4;
    const int vox = blockVox + nw * 16 + n;
    const bool doMath = (mTile * 16 < Co);   // wave-uniform

    v8f acc = {};

    for (int t = 0; t < 27; ++t) {
        __syncthreads();
        if (tid < nSlots) {
            int vv = blockVox + tid;
            int si = -1;
            if (vv < N) {
                int w = vv % Wd, h = (vv / Wd) % H, d = vv / (Wd * H);
                int kd = t / 9, kh = (t / 3) % 3, kw = t % 3;
                int sd = d + kd - 1, sh = h + kh - 1, sw = w + kw - 1;
                if (sd >= 0 && sd < D && sh >= 0 && sh < H && sw >= 0 && sw < Wd)
                    si = (sd * H + sh) * Wd + sw;
            }
            s_src[tid] = si;
        }
        __syncthreads();
        for (int i = tid; i < nSlots * CiP; i += blockDim.x) {
            int s = i % nSlots;
            int cip = i / nSlots;
            int si = s_src[s];
            _Float16 val = (_Float16)0.f;
            if (cip < Ci && si >= 0) val = act[(long)cip * N + si];
            sm[s * ldk + cip] = val;
        }
        __syncthreads();
        if (doMath) {
            const _Float16* aRow = W + ((long)(t * Co + mTile * 16 + n)) * CiP;
            if (t < 26) __builtin_prefetch(aRow + (long)Co * CiP, 0, 1);
            const _Float16* bRow = sm + (nw * 16 + n) * ldk;
            for (int kb = 0; kb < CiP; kb += 32) {
                v16h a = load_a_frag(aRow, g, kb);
                v16h b = load_b_frag(bRow, g, kb);
                acc = __builtin_amdgcn_wmma_f32_16x16x32_f16(false, a, false, b,
                                                             (short)0, acc, false, false);
            }
        }
    }

    if (doMath && vox < N) {
        int w = vox % Wd, h = (vox / Wd) % H, d = vox / (Wd * H);
        float supv = 1.f;
        if (sup) {
            int pidx = ((d >> 1) * (H >> 1) + (h >> 1)) * (Wd >> 1) + (w >> 1);
            supv = sup[pidx] ? 1.f : 0.f;
        }
#pragma unroll
        for (int r = 0; r < 8; ++r) {
            int co = mTile * 16 + r + 8 * g;
            float sc = bnG[co] * rsqrtf(bnV[co] + BN_EPS);
            float x = (acc[r] - bnM[co]) * sc + bnB[co];
            out[(long)co * N + vox] = (_Float16)(elu1(x) * supv);
        }
    }
}

// 1x1x1 classifier head: cls[v] = b + sum_c w[c]*act[c][v]  (f32 out into d_out)
__global__ void k_cls(const _Float16* __restrict__ act, const float* __restrict__ w,
                      const float* __restrict__ bias, float* __restrict__ outp,
                      int C, long N) {
    long v = blockIdx.x * (long)blockDim.x + threadIdx.x;
    if (v >= N) return;
    float s = bias[0];
    for (int c = 0; c < C; ++c) s += (float)act[(long)c * N + v] * w[c];
    outp[v] = s;
}

// keep = (cls>0) && parent-keep (at half res); stage-1 has no parent.
__global__ void k_keep(const float* __restrict__ cls, const unsigned char* __restrict__ kp,
                       unsigned char* __restrict__ keep, int R) {
    long N = (long)R * R * R;
    long v = blockIdx.x * (long)blockDim.x + threadIdx.x;
    if (v >= N) return;
    int w = (int)(v % R), h = (int)((v / R) % R), d = (int)(v / ((long)R * R));
    int parent = 1;
    if (kp) parent = kp[((d >> 1) * (R >> 1) + (h >> 1)) * (R >> 1) + (w >> 1)] ? 1 : 0;
    keep[v] = (cls[v] > 0.f && parent) ? 1 : 0;
}

// tgt = (maxpool(occ, f) > 0) && parent-keep ; occ is 128^3 bool.
__global__ void k_target(const int* __restrict__ occ, const unsigned char* __restrict__ kp,
                         float* __restrict__ tgt, int R, int f) {
    long N = (long)R * R * R;
    long v = blockIdx.x * (long)blockDim.x + threadIdx.x;
    if (v >= N) return;
    int w = (int)(v % R), h = (int)((v / R) % R), d = (int)(v / ((long)R * R));
    int any = 0;
    for (int zz = 0; zz < f && !any; ++zz)
        for (int hh = 0; hh < f && !any; ++hh)
            for (int ww = 0; ww < f && !any; ++ww) {
                long oi = ((long)(d * f + zz) * 128 + (h * f + hh)) * 128 + (w * f + ww);
                if (occ[oi] != 0) any = 1;
            }
    int parent = 1;
    if (kp) parent = kp[((d >> 1) * (R >> 1) + (h >> 1)) * (R >> 1) + (w >> 1)] ? 1 : 0;
    tgt[v] = (any && parent) ? 1.f : 0.f;
}

// act *= keep (broadcast over channels); optionally also emit f32 copy (final out).
__global__ void k_mask(_Float16* __restrict__ act, const unsigned char* __restrict__ keep,
                       float* __restrict__ fout, int C, long N) {
    long i = blockIdx.x * (long)blockDim.x + threadIdx.x;
    if (i >= (long)C * N) return;
    long v = i % N;
    float val = (float)act[i] * (keep[v] ? 1.f : 0.f);
    act[i] = (_Float16)val;
    if (fout) fout[i] = val;
}

// ---------------------------------------------------------------------------
extern "C" void kernel_launch(void* const* d_in, const int* in_sizes, int n_in,
                              void* d_out, int out_size, void* d_ws, size_t ws_size,
                              hipStream_t stream) {
    (void)in_sizes; (void)n_in; (void)out_size; (void)ws_size;
    const int CiA[6]  = {889, 1024, 512, 256, 128, 64};
    const int CoA[6]  = {1024, 512, 256, 128, 64, 32};
    const int DinA[6] = {2, 4, 8, 16, 32, 64};
    const int wdcI[6] = {1, 11, 23, 35, 47, 59};
    const int wcI[6]  = {6, 16, 28, 40, 52, 64};
    const int clsI[5] = {21, 33, 45, 57, 69};
    auto F = [&](int i) { return (const float*)d_in[i]; };

    // workspace layout
    size_t cur = 0;
    auto alloc = [&](size_t bytes) { size_t o = cur; cur += (bytes + 255) & ~(size_t)255; return o; };
    size_t wdcOff[6], wcOff[6];
    int CiP[6];
    for (int s = 0; s < 6; ++s) {
        CiP[s] = (CiA[s] + 31) / 32 * 32;
        wdcOff[s] = alloc((size_t)8 * CoA[s] * CiP[s] * 2);
        wcOff[s]  = alloc((size_t)27 * CoA[s] * CoA[s] * 2);
    }
    size_t zOff   = alloc((size_t)889 * 8 * 2);
    size_t bufAOff = alloc((size_t)32 * 2097152 * 2);   // max deconv out (f16)
    size_t bufBOff = alloc((size_t)32 * 2097152 * 2);   // max conv out (f16)
    size_t keepOff[5];
    for (int i = 0; i < 5; ++i) { int R = 8 << i; keepOff[i] = alloc((size_t)R * R * R); }
    char* ws = (char*)d_ws;

    // d_out layout: cls1..5 | tgt1..5 | final out
    const long clsOffs[5] = {0, 512, 4608, 37376, 299520};
    const long tgtBase = 2396672;
    const long finalOff = 4793344;

    // weight + input conversion
    for (int s = 0; s < 6; ++s) {
        k_cvt_w<<<dim3(1024), dim3(256), 0, stream>>>(F(wdcI[s]), (_Float16*)(ws + wdcOff[s]),
                                                      CoA[s], CiA[s], CiP[s], 8);
        k_cvt_w<<<dim3(2048), dim3(256), 0, stream>>>(F(wcI[s]), (_Float16*)(ws + wcOff[s]),
                                                      CoA[s], CoA[s], CoA[s], 27);
    }
    k_cvt_act<<<dim3(28), dim3(256), 0, stream>>>(F(0), (_Float16*)(ws + zOff), (long)889 * 8);

    const _Float16* actIn = (const _Float16*)(ws + zOff);
    for (int s = 0; s < 6; ++s) {
        int mTiles = CoA[s] / 16;
        int mW = mTiles < 8 ? mTiles : 8;
        int nW = 8 / mW;
        int Din = DinA[s], Nin = Din * Din * Din;
        int Dout = Din * 2;
        long Nout = (long)Dout * Dout * Dout;
        _Float16* dco = (_Float16*)(ws + bufAOff);
        _Float16* cco = (_Float16*)(ws + bufBOff);

        {   // deconv (BN1 + ELU fused)
            int nTiles = (Nin + 15) / 16;
            dim3 grid((nTiles + nW - 1) / nW, (mTiles + mW - 1) / mW);
            size_t sh = (size_t)nW * 16 * (CiP[s] + 8) * 2;
            k_deconv<<<grid, 256, sh, stream>>>(actIn, (const _Float16*)(ws + wdcOff[s]),
                F(wdcI[s] + 1), F(wdcI[s] + 2), F(wdcI[s] + 3), F(wdcI[s] + 4),
                dco, CiA[s], CiP[s], CoA[s], Din, Din, Din, mW, nW);
        }
        {   // conv3 (BN2 + ELU + support mask fused)
            int nTiles = (int)((Nout + 15) / 16);
            dim3 grid((nTiles + nW - 1) / nW, (mTiles + mW - 1) / mW);
            size_t sh = (size_t)nW * 16 * (CoA[s] + 8) * 2;
            const unsigned char* sup = (s >= 2) ? (const unsigned char*)(ws + keepOff[s - 2]) : nullptr;
            k_conv3<<<grid, 256, sh, stream>>>(dco, (const _Float16*)(ws + wcOff[s]),
                F(wcI[s] + 1), F(wcI[s] + 2), F(wcI[s] + 3), F(wcI[s] + 4),
                sup, cco, CoA[s], CoA[s], CoA[s], Dout, Dout, Dout, mW, nW);
        }
        if (s >= 1) {
            int i = s - 1;                      // cls/target/keep index 0..4
            int R = Dout;
            long N = Nout;
            float* clsOut = (float*)d_out + clsOffs[i];
            k_cls<<<dim3((unsigned)((N + 255) / 256)), 256, 0, stream>>>(
                cco, F(clsI[i]), F(clsI[i] + 1), clsOut, CoA[s], N);
            const unsigned char* kp = (i >= 1) ? (const unsigned char*)(ws + keepOff[i - 1]) : nullptr;
            int f = 128 / R;
            k_target<<<dim3((unsigned)((N + 255) / 256)), 256, 0, stream>>>(
                (const int*)d_in[71], kp, (float*)d_out + tgtBase + clsOffs[i], R, f);
            unsigned char* kcur = (unsigned char*)(ws + keepOff[i]);
            k_keep<<<dim3((unsigned)((N + 255) / 256)), 256, 0, stream>>>(clsOut, kp, kcur, R);
            float* fout = (s == 5) ? (float*)d_out + finalOff : nullptr;
            long CN = (long)CoA[s] * N;
            k_mask<<<dim3((unsigned)((CN + 255) / 256)), 256, 0, stream>>>(cco, kcur, fout, CoA[s], N);
        }
        actIn = cco;
    }
}
